// GINES_2113123910402
// MI455X (gfx1250) — matrix operand order repaired and checked by reading, not compiled, not run on hardware
//
#include <hip/hip_runtime.h>
#include <math.h>

#define HD 64
#define NN 100000
#define EE 1000000
#define GG 100

#define ETILES 8   // edge tiles (128 edges each) per block
#define NTILES 4   // node tiles (128 rows each) per block

typedef __bf16 bf16;
typedef __attribute__((ext_vector_type(16))) __bf16 v16bf;
typedef __attribute__((ext_vector_type(8)))  float  v8f;

__device__ __forceinline__ float reluf(float x) { return x > 0.f ? x : 0.f; }

// Async global->LDS copy of one DWORD per lane (GLOBAL_LOAD_ASYNC_TO_LDS_B32,
// GV mode: per-lane 64-bit global address, per-lane 32-bit LDS byte offset).
// Tracked by ASYNCcnt; pair with s_wait_asynccnt before the block barrier.
__device__ __forceinline__ void async_copy_b32(void* lds_ptr, const void* gptr) {
    uint32_t l = (uint32_t)(uintptr_t)lds_ptr;       // low 32 bits = LDS byte offset
    uint64_t g = (uint64_t)(uintptr_t)gptr;
    asm volatile("global_load_async_to_lds_b32 %0, %1, off"
                 :: "v"(l), "v"(g) : "memory");
}
__device__ __forceinline__ void wait_async0() {
    asm volatile("s_wait_asynccnt 0x0" ::: "memory");
}

// ---------------------------------------------------------------------------
// Edge message + scatter for conv layers with cin=64:
//   msg[e,:] = relu((h[src]-h[dst]) @ W + l2_b + edge_attr[e]*le_w + le_b)
//   agg[dst[e],:] += msg[e,:]
// Persistent block: weights staged once, then ETILES x 128 edges processed.
// ---------------------------------------------------------------------------
__global__ __launch_bounds__(256)
void edge_msg_wmma(const bf16* __restrict__ hb,
                   const float* __restrict__ ea,
                   const int* __restrict__ src,
                   const int* __restrict__ dst,
                   const float* __restrict__ l2w,   // [64][64] (k,n) row-major
                   const float* __restrict__ l2b,
                   const float* __restrict__ lew,   // [64]
                   const float* __restrict__ leb,   // [64]
                   float* __restrict__ agg,
                   int E)
{
    __shared__ alignas(32) bf16 sW[8][32][16];   // pre-swizzled B fragments
    __shared__ alignas(16) bf16 sD[128][72];     // diff tile, padded stride
    __shared__ float sBias[64];
    __shared__ float sLew[64];
    __shared__ float sEa[128];
    __shared__ int   sDst[128];

    const int t    = threadIdx.x;
    const int lane = t & 31;
    const int wave = t >> 5;
    const int e_base = blockIdx.x * (128 * ETILES);

    // Stage B fragments of W (bf16) once, exact 32x16 ISA layout per fragment.
    {
        const int frag = wave;                 // 4 n-tiles x 2 k-steps
        const int nt = frag >> 1, ks = frag & 1;
        const int n  = nt * 16 + (lane & 15);
        const int kbw = ks * 32 + ((lane >> 4) << 3);
        #pragma unroll
        for (int i = 0; i < 16; ++i)
            sW[frag][lane][i] = (bf16)l2w[(kbw + i + (i & 8)) * HD + n];
    }
    if (t < 64) { sBias[t] = l2b[t] + leb[t]; sLew[t] = lew[t]; }
    __syncthreads();

    const int m  = lane & 15;
    const int hi = lane >> 4;
    const int kb = hi * 8;
    const int arow = wave * 16 + m;

    float biasv[4], lewv[4];
    #pragma unroll
    for (int nt = 0; nt < 4; ++nt) { biasv[nt] = sBias[nt * 16 + m]; lewv[nt] = sLew[nt * 16 + m]; }

    for (int it = 0; it < ETILES; ++it) {
        const int e0 = e_base + it * 128;     // block-uniform
        if (e0 < E) {
            // Async-stage edge scalars (pure contiguous copies -> ASYNC DMA path).
            if (t < 128) {
                int e = e0 + t; e = e < E ? e : E - 1;
                async_copy_b32(&sEa[t], ea + e);
                async_copy_b32(&sDst[t], dst + e);
            }
            // Stage diff tile: 2 threads per edge row, 32 feats each (bf16 L2 gather).
            const int row = t >> 1, half = t & 1, fb = half * 32;
            const int e = e0 + row;
            if (e < E) {
                const bf16* ps = hb + (size_t)src[e] * HD + fb;
                const bf16* pd = hb + (size_t)dst[e] * HD + fb;
                #pragma unroll
                for (int f = 0; f < 32; ++f)
                    sD[row][fb + f] = (bf16)((float)ps[f] - (float)pd[f]);
            } else {
                #pragma unroll
                for (int f = 0; f < 32; ++f) sD[row][fb + f] = (bf16)0.f;
            }
        }
        wait_async0();
        __syncthreads();

        if (e0 < E) {
            // A fragments (16x32 bf16 ISA layout).
            v16bf a0, a1;
            #pragma unroll
            for (int i = 0; i < 8; ++i) {
                a0[i]     = sD[arow][kb + i];
                a0[8 + i] = sD[arow][kb + 16 + i];
                a1[i]     = sD[arow][32 + kb + i];
                a1[8 + i] = sD[arow][32 + kb + 16 + i];
            }
            // All 4 output tiles (A fragments reused across n-tiles).
            v8f acc[4];
            #pragma unroll
            for (int nt = 0; nt < 4; ++nt) {
                v8f c = {};
                v16bf b0 = *(const v16bf*)&sW[nt * 2 + 0][lane][0];
                v16bf b1 = *(const v16bf*)&sW[nt * 2 + 1][lane][0];
                c = __builtin_amdgcn_wmma_f32_16x16x32_bf16(false, a0, false, b0, (short)0, c, false, false);
                c = __builtin_amdgcn_wmma_f32_16x16x32_bf16(false, a1, false, b1, (short)0, c, false, false);
                acc[nt] = c;
            }
            // Row-major epilogue: one guard / one dst base per output row, then 4
            // atomics at immediate offsets off one address pair.
            #pragma unroll
            for (int r = 0; r < 8; ++r) {
                const int er = wave * 16 + r + (hi << 3);   // C layout: M = r + 8*(lane>=16)
                if (e0 + er < E) {
                    float* base = agg + (size_t)sDst[er] * HD + m;
                    const float eav = sEa[er];
                    #pragma unroll
                    for (int nt = 0; nt < 4; ++nt) {
                        float v = reluf(acc[nt][r] + biasv[nt] + eav * lewv[nt]);
                        atomicAdd(base + nt * 16, v);
                    }
                }
            }
        }
        __syncthreads();   // protect sD/sEa/sDst before next tile's staging
    }
}

// ---------------------------------------------------------------------------
// Node stage 1 (cin=64): mid = relu(bn((agg+h) @ n1_w + n1_b))  -> bf16
// BatchNorm folded into per-column scale/bias. Persistent over NTILES tiles.
// ---------------------------------------------------------------------------
__global__ __launch_bounds__(256)
void node_stage1_wmma(const float* __restrict__ agg,
                      const float* __restrict__ h,
                      const float* __restrict__ w1,
                      const float* __restrict__ b1,
                      const float* __restrict__ bng, const float* __restrict__ bnb,
                      const float* __restrict__ bnm, const float* __restrict__ bnv,
                      bf16* __restrict__ mid, int N)
{
    __shared__ alignas(32) bf16 sW[8][32][16];
    __shared__ alignas(16) bf16 sIn[128][72];
    __shared__ float sScale[64], sBias[64];

    const int t = threadIdx.x, lane = t & 31, wave = t >> 5;
    const int r_base = blockIdx.x * (128 * NTILES);

    {
        const int frag = wave, nt = frag >> 1, ks = frag & 1;
        const int n  = nt * 16 + (lane & 15);
        const int kbw = ks * 32 + ((lane >> 4) << 3);
        #pragma unroll
        for (int i = 0; i < 16; ++i)
            sW[frag][lane][i] = (bf16)w1[(kbw + i + (i & 8)) * HD + n];
    }
    if (t < 64) {
        float sc = bng[t] * rsqrtf(bnv[t] + 1e-5f);
        sScale[t] = sc;
        sBias[t]  = (b1[t] - bnm[t]) * sc + bnb[t];
    }
    __syncthreads();

    const int m = lane & 15, hi = lane >> 4, kb = hi * 8;
    const int arow = wave * 16 + m;
    float scv[4], bsv[4];
    #pragma unroll
    for (int nt = 0; nt < 4; ++nt) { scv[nt] = sScale[nt * 16 + m]; bsv[nt] = sBias[nt * 16 + m]; }

    for (int it = 0; it < NTILES; ++it) {
        const int r0 = r_base + it * 128;     // block-uniform
        if (r0 < N) {
            const int row = t >> 1, half = t & 1, fb = half * 32;
            int gr = r0 + row; gr = gr < N ? gr : N - 1;
            const float* pa = agg + (size_t)gr * HD + fb;
            const float* ph = h   + (size_t)gr * HD + fb;
            #pragma unroll
            for (int f = 0; f < 32; ++f)
                sIn[row][fb + f] = (bf16)(pa[f] + ph[f]);
        }
        __syncthreads();

        if (r0 < N) {
            v16bf a0, a1;
            #pragma unroll
            for (int i = 0; i < 8; ++i) {
                a0[i]     = sIn[arow][kb + i];
                a0[8 + i] = sIn[arow][kb + 16 + i];
                a1[i]     = sIn[arow][32 + kb + i];
                a1[8 + i] = sIn[arow][32 + kb + 16 + i];
            }
            v8f acc[4];
            #pragma unroll
            for (int nt = 0; nt < 4; ++nt) {
                v8f c = {};
                v16bf b0 = *(const v16bf*)&sW[nt * 2 + 0][lane][0];
                v16bf b1 = *(const v16bf*)&sW[nt * 2 + 1][lane][0];
                c = __builtin_amdgcn_wmma_f32_16x16x32_bf16(false, a0, false, b0, (short)0, c, false, false);
                c = __builtin_amdgcn_wmma_f32_16x16x32_bf16(false, a1, false, b1, (short)0, c, false, false);
                acc[nt] = c;
            }
            #pragma unroll
            for (int r = 0; r < 8; ++r) {
                const int row = r0 + wave * 16 + r + (hi << 3);
                if (row < N) {
                    bf16* base = mid + (size_t)row * HD + m;
                    #pragma unroll
                    for (int nt = 0; nt < 4; ++nt)
                        base[nt * 16] = (bf16)reluf(acc[nt][r] * scv[nt] + bsv[nt]);
                }
            }
        }
        __syncthreads();   // protect sIn before next tile's staging
    }
}

// ---------------------------------------------------------------------------
// Node stage 2: z = relu(mid @ n2_w + n2_b) (f32 out) + PairNorm statistics
// (column sums + total sum of squares) via LDS ds_add_f32 block reduction.
// A-fragments loaded directly from global (rows contiguous) -> barrier-free loop.
// ---------------------------------------------------------------------------
__global__ __launch_bounds__(256)
void node_stage2_wmma(const bf16* __restrict__ mid,
                      const float* __restrict__ w2, const float* __restrict__ b2,
                      float* __restrict__ z,
                      float* __restrict__ colsum, float* __restrict__ sumsq,
                      int N)
{
    __shared__ alignas(32) bf16 sW[8][32][16];
    __shared__ float sB2[64];
    __shared__ float sCol[64];
    __shared__ float sSq;

    const int t = threadIdx.x, lane = t & 31, wave = t >> 5;
    const int r_base = blockIdx.x * (128 * NTILES);
    {
        const int frag = wave, nt = frag >> 1, ks = frag & 1;
        const int n  = nt * 16 + (lane & 15);
        const int kbw = ks * 32 + ((lane >> 4) << 3);
        #pragma unroll
        for (int i = 0; i < 16; ++i)
            sW[frag][lane][i] = (bf16)w2[(kbw + i + (i & 8)) * HD + n];
    }
    if (t < 64) { sB2[t] = b2[t]; sCol[t] = 0.f; }
    if (t == 64) sSq = 0.f;
    __syncthreads();

    const int m = lane & 15, hi = lane >> 4, kb = hi * 8;
    float bsv[4];
    #pragma unroll
    for (int nt = 0; nt < 4; ++nt) bsv[nt] = sB2[nt * 16 + m];

    float sq = 0.f;
    for (int it = 0; it < NTILES; ++it) {
        const int r0 = r_base + it * 128;
        if (r0 >= N) break;                    // block-uniform; no barriers in loop
        int arow = r0 + wave * 16 + m; arow = arow < N ? arow : N - 1;
        const bf16* pm = mid + (size_t)arow * HD;   // b128 global loads
        v16bf a0, a1;
        #pragma unroll
        for (int i = 0; i < 8; ++i) {
            a0[i]     = pm[kb + i];
            a0[8 + i] = pm[kb + 16 + i];
            a1[i]     = pm[32 + kb + i];
            a1[8 + i] = pm[32 + kb + 16 + i];
        }
        v8f acc[4];
        #pragma unroll
        for (int nt = 0; nt < 4; ++nt) {
            v8f c = {};
            v16bf b0 = *(const v16bf*)&sW[nt * 2 + 0][lane][0];
            v16bf b1 = *(const v16bf*)&sW[nt * 2 + 1][lane][0];
            c = __builtin_amdgcn_wmma_f32_16x16x32_bf16(false, a0, false, b0, (short)0, c, false, false);
            c = __builtin_amdgcn_wmma_f32_16x16x32_bf16(false, a1, false, b1, (short)0, c, false, false);
            acc[nt] = c;
        }
        #pragma unroll
        for (int r = 0; r < 8; ++r) {
            const int row = r0 + wave * 16 + r + (hi << 3);
            if (row < N) {
                float* base = z + (size_t)row * HD + m;
                #pragma unroll
                for (int nt = 0; nt < 4; ++nt) {
                    float v = reluf(acc[nt][r] + bsv[nt]);
                    base[nt * 16] = v;
                    atomicAdd(&sCol[nt * 16 + m], v);     // ds_add_f32
                    sq += v * v;
                }
            }
        }
    }
    atomicAdd(&sSq, sq);
    __syncthreads();
    if (t < 64) atomicAdd(&colsum[t], sCol[t]);
    if (t == 0) atomicAdd(sumsq, sSq);
}

// ---------------------------------------------------------------------------
// Layer-1 specials (cin = 2): scalar edge messages and 2->64 node lift.
// ---------------------------------------------------------------------------
__global__ void edge1_kernel(const float* __restrict__ x, const float* __restrict__ ea,
                             const int* __restrict__ src, const int* __restrict__ dst,
                             const float* __restrict__ w, const float* __restrict__ b,
                             const float* __restrict__ lw, const float* __restrict__ lb,
                             float* __restrict__ agg2, int E)
{
    int e = blockIdx.x * blockDim.x + threadIdx.x;
    if (e >= E) return;
    int s = src[e], d = dst[e];
    float d0 = x[2 * s] - x[2 * d], d1 = x[2 * s + 1] - x[2 * d + 1];
    float a = ea[e];
    float m0 = reluf(d0 * w[0] + d1 * w[2] + b[0] + a * lw[0] + lb[0]);
    float m1 = reluf(d0 * w[1] + d1 * w[3] + b[1] + a * lw[1] + lb[1]);
    atomicAdd(&agg2[(size_t)d * 2 + 0], m0);
    atomicAdd(&agg2[(size_t)d * 2 + 1], m1);
}

__global__ void node1_stage1(const float* __restrict__ agg2, const float* __restrict__ x,
                             const float* __restrict__ w1, const float* __restrict__ b1,
                             const float* __restrict__ bng, const float* __restrict__ bnb,
                             const float* __restrict__ bnm, const float* __restrict__ bnv,
                             bf16* __restrict__ mid, int N)
{
    int idx = blockIdx.x * blockDim.x + threadIdx.x;
    if (idx >= N * HD) return;
    int row = idx >> 6, n = idx & 63;
    float o0 = agg2[2 * row]     + x[2 * row];
    float o1 = agg2[2 * row + 1] + x[2 * row + 1];
    float v = o0 * w1[n] + o1 * w1[HD + n] + b1[n];
    v = (v - bnm[n]) * (bng[n] * rsqrtf(bnv[n] + 1e-5f)) + bnb[n];
    mid[idx] = (bf16)reluf(v);
}

// ---------------------------------------------------------------------------
// PairNorm finalize + apply (+ global_add_pool accumulation).
// ---------------------------------------------------------------------------
__global__ void pn_finalize(const float* __restrict__ colsum, const float* __restrict__ sumsq,
                            float* __restrict__ mean, float* __restrict__ scale, int N)
{
    __shared__ float ssum;
    int t = threadIdx.x;                       // 64 threads
    if (t == 0) ssum = 0.f;
    __syncthreads();
    float mn = colsum[t] / (float)N;
    mean[t] = mn;
    atomicAdd(&ssum, mn * mn);
    __syncthreads();
    if (t == 0) {
        float tot = sumsq[0] - (float)N * ssum;   // sum over (h - mean)^2
        scale[0] = rsqrtf(1e-6f + tot / (float)N);
    }
}

__global__ void pn_apply(const float* z, const float* __restrict__ mean,
                         const float* __restrict__ scale, const int* __restrict__ batch,
                         float* h, bf16* __restrict__ hb,
                         float* __restrict__ pool, int layerOff, int N)
{
    int idx = blockIdx.x * blockDim.x + threadIdx.x;
    if (idx >= N * HD) return;
    int row = idx >> 6, n = idx & 63;
    float v = (z[idx] - mean[n]) * scale[0];
    h[idx]  = v;
    hb[idx] = (bf16)v;
    atomicAdd(&pool[(size_t)batch[row] * (3 * HD) + layerOff + n], v);
}

// ---------------------------------------------------------------------------
// Head: [G,192] @ [192,192] + relu, then [G,192] @ [192,2] + log_softmax.
// ---------------------------------------------------------------------------
__global__ void head1(const float* __restrict__ pool, const float* __restrict__ w,
                      const float* __restrict__ b, float* __restrict__ g1)
{
    int idx = blockIdx.x * blockDim.x + threadIdx.x;
    if (idx >= GG * 192) return;
    int g = idx / 192, j = idx % 192;
    const float* pr = pool + (size_t)g * 192;
    float v = b[j];
    for (int k = 0; k < 192; ++k) v += pr[k] * w[k * 192 + j];
    g1[idx] = reluf(v);
}

__global__ void head2(const float* __restrict__ g1, const float* __restrict__ w,
                      const float* __restrict__ b, float* __restrict__ out)
{
    int g = blockIdx.x * blockDim.x + threadIdx.x;
    if (g >= GG) return;
    const float* pr = g1 + (size_t)g * 192;
    float s0 = b[0], s1 = b[1];
    for (int k = 0; k < 192; ++k) { s0 += pr[k] * w[2 * k]; s1 += pr[k] * w[2 * k + 1]; }
    float mx = fmaxf(s0, s1);
    float lse = mx + logf(expf(s0 - mx) + expf(s1 - mx));
    out[2 * g]     = s0 - lse;
    out[2 * g + 1] = s1 - lse;
}

// ---------------------------------------------------------------------------
extern "C" void kernel_launch(void* const* d_in, const int* in_sizes, int n_in,
                              void* d_out, int out_size, void* d_ws, size_t ws_size,
                              hipStream_t stream)
{
    (void)in_sizes; (void)n_in; (void)out_size; (void)ws_size;
    const float* x  = (const float*)d_in[0];
    const float* ea = (const float*)d_in[1];
    const float* P[3][12];   // le_w, le_b, l2_w, l2_b, n1_w, n1_b, bn_g, bn_b, bn_m, bn_v, n2_w, n2_b
    for (int l = 0; l < 3; ++l)
        for (int k = 0; k < 12; ++k)
            P[l][k] = (const float*)d_in[2 + l * 12 + k];
    const float* lin1w = (const float*)d_in[38];
    const float* lin1b = (const float*)d_in[39];
    const float* lin2w = (const float*)d_in[40];
    const float* lin2b = (const float*)d_in[41];
    const int* ei    = (const int*)d_in[42];
    const int* batch = (const int*)d_in[43];
    const int* src = ei;
    const int* dst = ei + EE;

    // Workspace layout (h doubles as z; hb doubles as mid).
    char* w = (char*)d_ws;
    size_t off = 0;
    auto alloc = [&](size_t bytes) { void* p = w + off; off += (bytes + 255) & ~(size_t)255; return p; };
    float* h     = (float*)alloc((size_t)NN * HD * 4);
    bf16*  hb    = (bf16*) alloc((size_t)NN * HD * 2);
    float* agg   = (float*)alloc((size_t)NN * HD * 4);
    float* pool  = (float*)alloc((size_t)GG * 3 * HD * 4);
    float* stats = (float*)alloc(256 * 4);
    float* g1    = (float*)alloc((size_t)GG * 192 * 4);
    float* colsum = stats;          // [0..63]
    float* sumsq  = stats + 64;     // [64]
    float* mean   = stats + 65;     // [65..128]
    float* scale  = stats + 129;    // [129]

    hipMemsetAsync(pool, 0, (size_t)GG * 3 * HD * 4, stream);

    const int eblk  = (EE + 128 * ETILES - 1) / (128 * ETILES);
    const int nblk  = (NN + 128 * NTILES - 1) / (128 * NTILES);
    const int nelem = NN * HD;

    // ---------------- conv1 (cin = 2) ----------------
    hipMemsetAsync(agg, 0, (size_t)NN * 2 * 4, stream);
    edge1_kernel<<<(EE + 255) / 256, 256, 0, stream>>>(x, ea, src, dst,
        P[0][2], P[0][3], P[0][0], P[0][1], agg, EE);
    node1_stage1<<<(nelem + 255) / 256, 256, 0, stream>>>(agg, x,
        P[0][4], P[0][5], P[0][6], P[0][7], P[0][8], P[0][9], hb, NN);
    hipMemsetAsync(stats, 0, 65 * 4, stream);
    node_stage2_wmma<<<nblk, 256, 0, stream>>>(hb, P[0][10], P[0][11], h, colsum, sumsq, NN);
    pn_finalize<<<1, 64, 0, stream>>>(colsum, sumsq, mean, scale, NN);
    pn_apply<<<(nelem + 255) / 256, 256, 0, stream>>>(h, mean, scale, batch, h, hb, pool, 0, NN);

    // ---------------- conv2 / conv3 (cin = 64, WMMA path) ----------------
    for (int l = 1; l < 3; ++l) {
        hipMemsetAsync(agg, 0, (size_t)NN * HD * 4, stream);
        edge_msg_wmma<<<eblk, 256, 0, stream>>>(hb, ea, src, dst,
            P[l][2], P[l][3], P[l][0], P[l][1], agg, EE);
        node_stage1_wmma<<<nblk, 256, 0, stream>>>(agg, h,
            P[l][4], P[l][5], P[l][6], P[l][7], P[l][8], P[l][9], hb, NN);
        hipMemsetAsync(stats, 0, 65 * 4, stream);
        node_stage2_wmma<<<nblk, 256, 0, stream>>>(hb, P[l][10], P[l][11], h, colsum, sumsq, NN);
        pn_finalize<<<1, 64, 0, stream>>>(colsum, sumsq, mean, scale, NN);
        pn_apply<<<(nelem + 255) / 256, 256, 0, stream>>>(h, mean, scale, batch, h, hb, pool, l * HD, NN);
    }

    // ---------------- head ----------------
    head1<<<(GG * 192 + 255) / 256, 256, 0, stream>>>(pool, lin1w, lin1b, g1);
    head2<<<1, 128, 0, stream>>>(g1, lin2w, lin2b, (float*)d_out);
}
